// AMemNetModel_41369124995116
// MI455X (gfx1250) — matrix operand order repaired and checked by reading, not compiled, not run on hardware
//
#include <hip/hip_runtime.h>
#include <hip/hip_bf16.h>

// ---------------- problem constants ----------------
constexpr int B_   = 64;
constexpr int D_   = 2048;
constexpr int AV_  = 512;
constexpr int AN_  = 196;   // 14*14
constexpr int H_   = 1024;
constexpr int RG_  = 512;
constexpr int T_   = 8;     // STEPS

typedef __attribute__((ext_vector_type(16))) _Float16 v16h;
typedef __attribute__((ext_vector_type(8)))  _Float16 h8;
typedef __attribute__((ext_vector_type(8)))  float    v8f;
typedef __attribute__((ext_vector_type(4)))  float    f4;

// ---------------- WMMA fragment builders -------------
// A-matrix 16x32 f16 (k order chosen here, not the ISA table order): lane = m + 16*half; lane's 16 elements are
// K = {16h..16h+15}  (16 consecutive K values, matching the B fragments).
__device__ __forceinline__ v16h make_a_frag_f16(const _Float16* rowK, int half) {
  h8 a = *(const h8*)(rowK + 16 * half);
  h8 b = *(const h8*)(rowK + 16 * half + 8);
  v16h f;
#pragma unroll
  for (int i = 0; i < 8; ++i) { f[i] = a[i]; f[8 + i] = b[i]; }
  return f;
}
__device__ __forceinline__ v16h make_a_frag_f32(const float* rowK, int half) {
  f4 a0 = *(const f4*)(rowK + 16 * half);
  f4 a1 = *(const f4*)(rowK + 16 * half + 4);
  f4 b0 = *(const f4*)(rowK + 16 * half + 8);
  f4 b1 = *(const f4*)(rowK + 16 * half + 12);
  v16h f;
#pragma unroll
  for (int i = 0; i < 4; ++i) {
    f[i]      = (_Float16)a0[i];
    f[4 + i]  = (_Float16)a1[i];
    f[8 + i]  = (_Float16)b0[i];
    f[12 + i] = (_Float16)b1[i];
  }
  return f;
}
// B-matrix 32x16 f16: lane = n + 16*khalf; 16 consecutive K values.
__device__ __forceinline__ v16h make_b_frag_f32(const float* p /* &W[n][k0+khalf*16] */) {
  v16h f;
#pragma unroll
  for (int i = 0; i < 16; i += 4) {
    f4 a = *(const f4*)(p + i);
    f[i] = (_Float16)a[0]; f[i + 1] = (_Float16)a[1];
    f[i + 2] = (_Float16)a[2]; f[i + 3] = (_Float16)a[3];
  }
  return f;
}

__device__ __forceinline__ v8f wmma_(v16h a, v16h b, v8f c) {
  return __builtin_amdgcn_wmma_f32_16x16x32_f16(false, a, false, b, (short)0, c, false, false);
}

__device__ __forceinline__ float sigmoidf_(float x) { return 1.0f / (1.0f + expf(-x)); }

// ---------------- pack kernels ----------------
__global__ void pack_f16_kernel(const float* __restrict__ src, _Float16* __restrict__ dst, int n) {
  int i = blockIdx.x * 256 + threadIdx.x;
  if (i < n) dst[i] = (_Float16)src[i];
}

// xT[b][n][d] = (f16) x[b][d][n] ; LDS tile transpose, coalesced both sides.
__global__ __launch_bounds__(256) void packxT_kernel(const float* __restrict__ x,
                                                     _Float16* __restrict__ xT) {
  __shared__ _Float16 tile[64][AN_ + 4];
  const int b  = blockIdx.y;
  const int d0 = blockIdx.x * 64;
  for (int i = threadIdx.x; i < 64 * AN_; i += 256) {
    int dd = i / AN_, n = i - dd * AN_;
    tile[dd][n] = (_Float16)x[((size_t)b * D_ + d0 + dd) * AN_ + n];
  }
  __syncthreads();
  for (int i = threadIdx.x; i < 64 * AN_; i += 256) {
    int n = i >> 6, dd = i & 63;
    xT[((size_t)b * AN_ + n) * D_ + d0 + dd] = tile[dd][n];
  }
}

// ---------------- inconv: a = relu(BN(w @ x + b)) ----------------
// LDS-free WMMA GEMM. grid: (ceil(AN/32)=7, B). block = 8 waves.
// wave: oBase = wave*64 -> MT=4 o-tiles; NT=2 n-tiles per block.
__global__ __launch_bounds__(256) void inconv_kernel(
    const _Float16* __restrict__ wh,    // [AV, D] f16
    const _Float16* __restrict__ xT,    // [B, AN, D] f16
    const float* __restrict__ bias, const float* __restrict__ gamma,
    const float* __restrict__ beta, const float* __restrict__ mean,
    const float* __restrict__ var,
    _Float16* __restrict__ a16T)        // [B, AN, AV]
{
  const int lane  = threadIdx.x & 31;
  const int wave  = threadIdx.x >> 5;
  const int half  = lane >> 4;
  const int ln    = lane & 15;
  const int b     = blockIdx.y;
  const int nBase = blockIdx.x * 32;
  const int oBase = wave * 64;

  const _Float16* brow[2];
#pragma unroll
  for (int nt = 0; nt < 2; ++nt) {
    int n = nBase + nt * 16 + ln;
    n = (n < AN_) ? n : (AN_ - 1);
    brow[nt] = xT + ((size_t)b * AN_ + n) * D_ + half * 16;
  }
  const _Float16* arow[4];
#pragma unroll
  for (int mt = 0; mt < 4; ++mt)
    arow[mt] = wh + (size_t)(oBase + mt * 16 + ln) * D_;

  v8f acc[4][2] = {};
  for (int k0 = 0; k0 < D_; k0 += 32) {
    // preload all fragments so the scheduler can overlap loads with WMMAs
    v16h fb0 = *(const v16h*)(brow[0] + k0);
    v16h fb1 = *(const v16h*)(brow[1] + k0);
    v16h fa[4];
#pragma unroll
    for (int mt = 0; mt < 4; ++mt)
      fa[mt] = make_a_frag_f16(arow[mt] + k0, half);
    if (k0 + 256 < D_) __builtin_prefetch(brow[0] + k0 + 256, 0, 3);
#pragma unroll
    for (int mt = 0; mt < 4; ++mt) {
      acc[mt][0] = wmma_(fa[mt], fb0, acc[mt][0]);
      acc[mt][1] = wmma_(fa[mt], fb1, acc[mt][1]);
    }
  }

#pragma unroll
  for (int mt = 0; mt < 4; ++mt) {
#pragma unroll
    for (int r = 0; r < 8; ++r) {
      int o = oBase + mt * 16 + r + 8 * half;
      float sc = gamma[o] * rsqrtf(var[o] + 1e-5f);
      float sh = beta[o] - mean[o] * sc;
#pragma unroll
      for (int nt = 0; nt < 2; ++nt) {
        int n = nBase + nt * 16 + ln;
        if (n < AN_) {
          float v = fmaxf((acc[mt][nt][r] + bias[o]) * sc + sh, 0.0f);
          a16T[((size_t)b * AN_ + n) * AV_ + o] = (_Float16)v;
        }
      }
    }
  }
}

// ---------------- eproj: e = relu(e1_w @ a) -> f16 [B, AV, AN] ----------------
__global__ __launch_bounds__(256) void eproj_kernel(
    const _Float16* __restrict__ e1h,   // [AV, AV] f16
    const _Float16* __restrict__ a16T,  // [B, AN, AV] f16
    _Float16* __restrict__ e16)         // [B, AV, AN]
{
  const int lane  = threadIdx.x & 31;
  const int wave  = threadIdx.x >> 5;
  const int half  = lane >> 4;
  const int ln    = lane & 15;
  const int b     = blockIdx.y;
  const int nBase = blockIdx.x * 32;
  const int oBase = wave * 64;

  const _Float16* brow[2];
#pragma unroll
  for (int nt = 0; nt < 2; ++nt) {
    int n = nBase + nt * 16 + ln;
    n = (n < AN_) ? n : (AN_ - 1);
    brow[nt] = a16T + ((size_t)b * AN_ + n) * AV_ + half * 16;
  }
  const _Float16* arow[4];
#pragma unroll
  for (int mt = 0; mt < 4; ++mt)
    arow[mt] = e1h + (size_t)(oBase + mt * 16 + ln) * AV_;

  v8f acc[4][2] = {};
  for (int k0 = 0; k0 < AV_; k0 += 32) {
    v16h fb0 = *(const v16h*)(brow[0] + k0);
    v16h fb1 = *(const v16h*)(brow[1] + k0);
    v16h fa[4];
#pragma unroll
    for (int mt = 0; mt < 4; ++mt)
      fa[mt] = make_a_frag_f16(arow[mt] + k0, half);
#pragma unroll
    for (int mt = 0; mt < 4; ++mt) {
      acc[mt][0] = wmma_(fa[mt], fb0, acc[mt][0]);
      acc[mt][1] = wmma_(fa[mt], fb1, acc[mt][1]);
    }
  }

#pragma unroll
  for (int mt = 0; mt < 4; ++mt)
#pragma unroll
    for (int r = 0; r < 8; ++r) {
      int o = oBase + mt * 16 + r + 8 * half;
#pragma unroll
      for (int nt = 0; nt < 2; ++nt) {
        int n = nBase + nt * 16 + ln;
        if (n < AN_)
          e16[((size_t)b * AV_ + o) * AN_ + n] = (_Float16)fmaxf(acc[mt][nt][r], 0.0f);
      }
    }
}

// ---------------- generic dual-input linear (LDS-free WMMA) ----------------
// out[m,n] = act( A1@W1^T + A2@W2^T + b1 + b2 ), M == 64 (4 m-tiles per wave).
// grid: ceil(N/128). act: 0 none, 1 relu, 2 tanh
__global__ __launch_bounds__(256) void linear2_kernel(
    const float* __restrict__ A1, const float* __restrict__ W1, int K1,
    const float* __restrict__ A2, const float* __restrict__ W2, int K2,
    const float* __restrict__ bias1, const float* __restrict__ bias2,
    float* __restrict__ out, int N, int act)
{
  const int lane  = threadIdx.x & 31;
  const int wave  = threadIdx.x >> 5;
  const int half  = lane >> 4;
  const int ln    = lane & 15;
  const int nBase = blockIdx.x * 128 + wave * 16;
  const int n     = nBase + ln;
  const int nc    = (n < N) ? n : (N - 1);

  v8f acc[4] = {};
  for (int pass = 0; pass < 2; ++pass) {
    const float* A = pass ? A2 : A1;
    const float* W = pass ? W2 : W1;
    const int    K = pass ? K2 : K1;
    if (K == 0) continue;
    const float* wrow = W + (size_t)nc * K + half * 16;
    for (int k0 = 0; k0 < K; k0 += 32) {
      v16h fb = make_b_frag_f32(wrow + k0);
      v16h fa[4];
#pragma unroll
      for (int mt = 0; mt < 4; ++mt)
        fa[mt] = make_a_frag_f32(A + (size_t)(mt * 16 + ln) * K + k0, half);
#pragma unroll
      for (int mt = 0; mt < 4; ++mt)
        acc[mt] = wmma_(fa[mt], fb, acc[mt]);
    }
  }

  if (n < N) {
    float bv = (bias1 ? bias1[n] : 0.0f) + (bias2 ? bias2[n] : 0.0f);
#pragma unroll
    for (int mt = 0; mt < 4; ++mt)
#pragma unroll
      for (int r = 0; r < 8; ++r) {
        int m = mt * 16 + r + 8 * half;
        float v = acc[mt][r] + bv;
        if (act == 1) v = fmaxf(v, 0.0f);
        else if (act == 2) v = tanhf(v);
        out[(size_t)m * N + n] = v;
      }
  }
}

// ---------------- small pointwise / reduction kernels ----------------
// af[b,o] = mean_n a16T[b][n][o] ; coalesced over o. grid (AV/256, B)
__global__ void af_kernel(const _Float16* __restrict__ a16T, float* __restrict__ af) {
  int o = blockIdx.x * 256 + threadIdx.x;
  int b = blockIdx.y;
  const _Float16* p = a16T + (size_t)b * AN_ * AV_ + o;
  float s = 0.0f;
  for (int n = 0; n < AN_; ++n) s += (float)p[(size_t)n * AV_];
  af[b * AV_ + o] = s * (1.0f / AN_);
}

// logits + softmax fused; one block per batch.
__global__ __launch_bounds__(256) void attn_kernel(
    const _Float16* __restrict__ e16, const float* __restrict__ eh,
    const float* __restrict__ w3, float* __restrict__ alpha,
    float* __restrict__ out_alphas, int t)
{
  __shared__ float red[256];
  const int b = blockIdx.x;
  const int n = threadIdx.x;
  float logit = -1e30f;
  if (n < AN_) {
    float ehv = eh[b * AN_ + n];
    const _Float16* eb = e16 + (size_t)b * AV_ * AN_ + n;
    float acc = 0.0f;
    for (int v = 0; v < AV_; ++v)
      acc += w3[v] * fmaxf((float)eb[(size_t)v * AN_] + ehv, 0.0f);
    logit = acc;
  }
  red[threadIdx.x] = logit; __syncthreads();
  for (int s = 128; s > 0; s >>= 1) {
    if (threadIdx.x < s) red[threadIdx.x] = fmaxf(red[threadIdx.x], red[threadIdx.x + s]);
    __syncthreads();
  }
  float mx = red[0]; __syncthreads();
  float ex = (n < AN_) ? expf(logit - mx) : 0.0f;
  red[threadIdx.x] = ex; __syncthreads();
  for (int s = 128; s > 0; s >>= 1) {
    if (threadIdx.x < s) red[threadIdx.x] += red[threadIdx.x + s];
    __syncthreads();
  }
  float inv = 1.0f / red[0];
  if (n < AN_) {
    float al = ex * inv;
    alpha[b * AN_ + n] = al;
    out_alphas[((size_t)b * T_ + t) * AN_ + n] = al;
  }
}

// z[b,v] = sum_n a16T[b][n][v] * alpha[b][n] ; coalesced over v. grid (AV/256, B)
__global__ void z_kernel(const _Float16* __restrict__ a16T, const float* __restrict__ alpha,
                         float* __restrict__ z) {
  int v = blockIdx.x * 256 + threadIdx.x;
  int b = blockIdx.y;
  const _Float16* p = a16T + (size_t)b * AN_ * AV_ + v;
  const float* al = alpha + b * AN_;
  float s = 0.0f;
  for (int n = 0; n < AN_; ++n) s += (float)p[(size_t)n * AV_] * al[n];
  z[b * AV_ + v] = s;
}

__global__ void lstm_kernel(const float* __restrict__ gates,
                            float* __restrict__ h, float* __restrict__ c) {
  int i = blockIdx.x * blockDim.x + threadIdx.x;     // b*H + j
  if (i >= B_ * H_) return;
  int b = i / H_, j = i - b * H_;
  const float* g = gates + (size_t)b * 4 * H_;
  float ig = g[j], fg = g[H_ + j], gg = g[2 * H_ + j], og = g[3 * H_ + j];
  float cv = sigmoidf_(fg) * c[i] + sigmoidf_(ig) * tanhf(gg);
  float hv = sigmoidf_(og) * tanhf(cv);
  c[i] = cv;
  h[i] = hv;
}

__global__ void regdot_kernel(const float* __restrict__ r, const float* __restrict__ w4,
                              const float* __restrict__ b4, float* __restrict__ out, int t) {
  int wv   = (blockIdx.x * blockDim.x + threadIdx.x) >> 5;
  int lane = threadIdx.x & 31;
  if (wv >= B_) return;
  float acc = 0.0f;
  for (int j = lane; j < RG_; j += 32) acc += r[(size_t)wv * RG_ + j] * w4[j];
  for (int off = 16; off > 0; off >>= 1) acc += __shfl_xor(acc, off, 32);
  if (lane == 0) out[wv * T_ + t] = acc + b4[0];
}

// ---------------- host launch ----------------
extern "C" void kernel_launch(void* const* d_in, const int* in_sizes, int n_in,
                              void* d_out, int out_size, void* d_ws, size_t ws_size,
                              hipStream_t stream) {
  const float* x        = (const float*)d_in[0];
  const float* inconv_w = (const float*)d_in[1];
  const float* inconv_b = (const float*)d_in[2];
  const float* bn_gamma = (const float*)d_in[3];
  const float* bn_beta  = (const float*)d_in[4];
  const float* bn_mean  = (const float*)d_in[5];
  const float* bn_var   = (const float*)d_in[6];
  const float* hs1_w    = (const float*)d_in[7];
  const float* hs1_b    = (const float*)d_in[8];
  const float* hc1_w    = (const float*)d_in[9];
  const float* hc1_b    = (const float*)d_in[10];
  const float* e1_w     = (const float*)d_in[11];
  const float* eh1_w    = (const float*)d_in[12];
  const float* eh1_b    = (const float*)d_in[13];
  const float* eh3_w    = (const float*)d_in[14];
  const float* w_ih     = (const float*)d_in[15];
  const float* w_hh     = (const float*)d_in[16];
  const float* b_ih     = (const float*)d_in[17];
  const float* b_hh     = (const float*)d_in[18];
  const float* reg1_w   = (const float*)d_in[19];
  const float* reg1_b   = (const float*)d_in[20];
  const float* reg4_w   = (const float*)d_in[21];
  const float* reg4_b   = (const float*)d_in[22];

  char* wsp = (char*)d_ws;
  auto carve = [&](size_t bytes) {
    char* p = wsp;
    wsp += (bytes + 255) & ~(size_t)255;
    return p;
  };
  const size_t nA = (size_t)B_ * AV_ * AN_;
  _Float16* a16T  = (_Float16*)carve(nA * sizeof(_Float16));
  _Float16* e16   = (_Float16*)carve(nA * sizeof(_Float16));
  _Float16* xT    = (_Float16*)carve((size_t)B_ * AN_ * D_ * sizeof(_Float16));
  _Float16* wh    = (_Float16*)carve((size_t)AV_ * D_ * sizeof(_Float16));
  _Float16* e1h   = (_Float16*)carve((size_t)AV_ * AV_ * sizeof(_Float16));
  float*    af    = (float*)   carve((size_t)B_ * AV_ * sizeof(float));
  float*    h     = (float*)   carve((size_t)B_ * H_ * sizeof(float));
  float*    c     = (float*)   carve((size_t)B_ * H_ * sizeof(float));
  float*    eh    = (float*)   carve((size_t)B_ * AN_ * sizeof(float));
  float*    alpha = (float*)   carve((size_t)B_ * AN_ * sizeof(float));
  float*    z     = (float*)   carve((size_t)B_ * AV_ * sizeof(float));
  float*    gates = (float*)   carve((size_t)B_ * 4 * H_ * sizeof(float));
  float*    rbuf  = (float*)   carve((size_t)B_ * RG_ * sizeof(float));

  float* out        = (float*)d_out;          // [B, T] first
  float* out_alphas = out + (size_t)B_ * T_;  // then [B, T, AN]

  // 0) pack: xT (transposed f16), f16 weights for batch-reused GEMMs
  packxT_kernel<<<dim3(D_ / 64, B_), 256, 0, stream>>>(x, xT);
  pack_f16_kernel<<<(AV_ * D_ + 255) / 256, 256, 0, stream>>>(inconv_w, wh, AV_ * D_);
  pack_f16_kernel<<<(AV_ * AV_ + 255) / 256, 256, 0, stream>>>(e1_w, e1h, AV_ * AV_);

  // 1) inconv + BN + relu -> a16T
  inconv_kernel<<<dim3((AN_ + 31) / 32, B_), 256, 0, stream>>>(
      wh, xT, inconv_b, bn_gamma, bn_beta, bn_mean, bn_var, a16T);

  // 2) af = mean_n a
  af_kernel<<<dim3(AV_ / 256, B_), 256, 0, stream>>>(a16T, af);

  // 3) h0, c0
  linear2_kernel<<<H_ / 128, 256, 0, stream>>>(
      af, hs1_w, AV_, nullptr, nullptr, 0, hs1_b, nullptr, h, H_, 2);
  linear2_kernel<<<H_ / 128, 256, 0, stream>>>(
      af, hc1_w, AV_, nullptr, nullptr, 0, hc1_b, nullptr, c, H_, 2);

  // 4) e = relu(e1_w @ a)
  eproj_kernel<<<dim3((AN_ + 31) / 32, B_), 256, 0, stream>>>(e1h, a16T, e16);

  // 5) recurrent steps
  for (int t = 0; t < T_; ++t) {
    linear2_kernel<<<(AN_ + 127) / 128, 256, 0, stream>>>(
        h, eh1_w, H_, nullptr, nullptr, 0, eh1_b, nullptr, eh, AN_, 0);
    attn_kernel<<<B_, 256, 0, stream>>>(e16, eh, eh3_w, alpha, out_alphas, t);
    z_kernel<<<dim3(AV_ / 256, B_), 256, 0, stream>>>(a16T, alpha, z);
    linear2_kernel<<<(4 * H_) / 128, 256, 0, stream>>>(
        z, w_ih, AV_, h, w_hh, H_, b_ih, b_hh, gates, 4 * H_, 0);
    lstm_kernel<<<(B_ * H_ + 255) / 256, 256, 0, stream>>>(gates, h, c);
    linear2_kernel<<<RG_ / 128, 256, 0, stream>>>(
        h, reg1_w, H_, nullptr, nullptr, 0, reg1_b, nullptr, rbuf, RG_, 1);
    regdot_kernel<<<(B_ * 32 + 255) / 256, 256, 0, stream>>>(rbuf, reg4_w, reg4_b, out, t);
  }
}
